// ONLSTMStack_35510789603914
// MI455X (gfx1250) — matrix operand order repaired and checked by reading, not compiled, not run on hardware
//
#include <hip/hip_runtime.h>
#include <hip/hip_bf16.h>
#include <stdint.h>

// ---------------- problem constants ----------------
#define H_DIM   1024
#define CS_DIM  128
#define N_CH    8
#define G_DIM   4112          // 4*H + 2*N
#define B_DIM   64
#define L_SEQT  512
#define KC      2048          // fused K: [x_t | h_{t-1}]
#define BH      (B_DIM * H_DIM)   // 65536

typedef __bf16 bf16_t;
typedef __attribute__((ext_vector_type(16))) __bf16 v16bf;
typedef __attribute__((ext_vector_type(8)))  float  v8f;
typedef unsigned int u32;
typedef __attribute__((ext_vector_type(4))) u32 u32x4;

union FragU { v16bf v; u32x4 q[2]; };

// ---------------- setup kernels ----------------
__global__ void pack_weights_kernel(const float* __restrict__ Wih0, const float* __restrict__ Whh0,
                                    const float* __restrict__ Wih1, const float* __restrict__ Whh1,
                                    bf16_t* __restrict__ Wc)
{
    const size_t per_layer = (size_t)G_DIM * KC;
    size_t i = (size_t)blockIdx.x * blockDim.x + threadIdx.x;
    if (i >= 2 * per_layer) return;
    int layer = (int)(i / per_layer);
    size_t r = i - (size_t)layer * per_layer;
    int g = (int)(r / KC);
    int k = (int)(r - (size_t)g * KC);
    const float* Wih = layer ? Wih1 : Wih0;
    const float* Whh = layer ? Whh1 : Whh0;
    float v = (k < H_DIM) ? Wih[(size_t)g * H_DIM + k]
                          : Whh[(size_t)g * H_DIM + (k - H_DIM)];
    Wc[i] = (bf16_t)v;
}

__global__ void combine_bias_kernel(const float* __restrict__ bih0, const float* __restrict__ bhh0,
                                    const float* __restrict__ bih1, const float* __restrict__ bhh1,
                                    float* __restrict__ bias)
{
    int i = blockIdx.x * blockDim.x + threadIdx.x;
    if (i < G_DIM)            bias[i] = bih0[i] + bhh0[i];
    else if (i < 2 * G_DIM)   bias[i] = bih1[i - G_DIM] + bhh1[i - G_DIM];
}

__global__ void zero_u32_kernel(u32* __restrict__ p, int n)
{
    int i = blockIdx.x * blockDim.x + threadIdx.x;
    if (i < n) p[i] = 0u;
}

// ---------------- fragment loaders ----------------
// A fragment (16x32 bf16, M = batch row): lane l holds M=l%16 and two contiguous
// 8-element K runs at K = kA and K = kA+16 (kA = 8 for upper half-wave).
__device__ __forceinline__ v16bf load_a_bf(const bf16_t* __restrict__ ap)
{
    FragU t;
    t.q[0] = *(const u32x4*)ap;
    t.q[1] = *(const u32x4*)(ap + 16);
    return t.v;
}

__device__ __forceinline__ v16bf load_a_f32(const float* __restrict__ ap)
{
    float4 f0 = ((const float4*)ap)[0];
    float4 f1 = ((const float4*)ap)[1];
    float4 f2 = ((const float4*)(ap + 16))[0];
    float4 f3 = ((const float4*)(ap + 16))[1];
    v16bf av;
    av[0]=(bf16_t)f0.x;  av[1]=(bf16_t)f0.y;  av[2]=(bf16_t)f0.z;  av[3]=(bf16_t)f0.w;
    av[4]=(bf16_t)f1.x;  av[5]=(bf16_t)f1.y;  av[6]=(bf16_t)f1.z;  av[7]=(bf16_t)f1.w;
    av[8]=(bf16_t)f2.x;  av[9]=(bf16_t)f2.y;  av[10]=(bf16_t)f2.z; av[11]=(bf16_t)f2.w;
    av[12]=(bf16_t)f3.x; av[13]=(bf16_t)f3.y; av[14]=(bf16_t)f3.z; av[15]=(bf16_t)f3.w;
    return av;
}

// ---------------- fused projection GEMM: g = [x_t|h] @ Wc^T + bias ----------------
// Wave layout: each wave owns M=64 (4 tiles) x N=32 (2 tiles); block = 4 waves (N=128).
// B fragment (32x16 bf16): lane l holds N=l%16, two 8-elem K runs at 8*hi and 16+8*hi -> direct
//   b128 loads from row-major Wc[g][k]; no transpose needed.
// Templated on A source type so the two K-halves are branch-free straight lines.
template <bool A_IS_F32>
__global__ __launch_bounds__(128)
void onlstm_gemm_step(const void*   __restrict__ xin,       // 64x1024 fp32 or bf16
                      const bf16_t* __restrict__ hprev,     // 64x1024 bf16
                      const bf16_t* __restrict__ Wc,        // G_DIM x KC bf16, row-major
                      const float*  __restrict__ bias,      // G_DIM fp32
                      float*        __restrict__ gout)      // 64 x G_DIM fp32
{
    const int lane    = threadIdx.x & 31;
    const int wave    = threadIdx.x >> 5;
    const int colbase = blockIdx.x * 128 + wave * 32;
    if (colbase >= G_DIM) return;                 // wave-uniform exit (EXEC stays all-ones)

    const int mrow  = lane & 15;                  // M (A) / N (B) sub-index
    const int hi    = lane >> 4;                  // half-wave select
    const int g0    = colbase + mrow;             // N column, tile 0 (always valid)
    const int g1raw = colbase + 16 + mrow;        // N column, tile 1
    const int g1    = (g1raw < G_DIM) ? g1raw : (G_DIM - 1);   // clamp loads

    const bf16_t* b0p = Wc + (size_t)g0 * KC + hi * 8;
    const bf16_t* b1p = Wc + (size_t)g1 * KC + hi * 8;
    const int kA = hi * 8;

    v8f acc[4][2] = {};

    // ---- K half 1: A from xin (K tiles 0..31), branch-free ----
    {
        const float*  axf = (const float*)xin;
        const bf16_t* axb = (const bf16_t*)xin;
        #pragma unroll 2
        for (int kt = 0; kt < 32; ++kt) {
            const int k = kt * 32;
            FragU b0, b1;
            b0.q[0] = *(const u32x4*)(b0p + k);
            b0.q[1] = *(const u32x4*)(b0p + k + 16);
            b1.q[0] = *(const u32x4*)(b1p + k);
            b1.q[1] = *(const u32x4*)(b1p + k + 16);

            v16bf a[4];
            #pragma unroll
            for (int mt = 0; mt < 4; ++mt) {
                const size_t aoff = (size_t)(mt * 16 + mrow) * H_DIM + k + kA;
                if (A_IS_F32) a[mt] = load_a_f32(axf + aoff);
                else          a[mt] = load_a_bf (axb + aoff);
            }
            #pragma unroll
            for (int mt = 0; mt < 4; ++mt) {
                acc[mt][0] = __builtin_amdgcn_wmma_f32_16x16x32_bf16(
                    false, a[mt], false, b0.v, (short)0, acc[mt][0], false, false);
                acc[mt][1] = __builtin_amdgcn_wmma_f32_16x16x32_bf16(
                    false, a[mt], false, b1.v, (short)0, acc[mt][1], false, false);
            }
        }
    }

    // ---- K half 2: A from hprev (K tiles 32..63), branch-free ----
    {
        #pragma unroll 2
        for (int kt = 0; kt < 32; ++kt) {
            const int k = kt * 32;               // offset within hprev
            const int kw = 1024 + k;             // offset within Wc row
            FragU b0, b1;
            b0.q[0] = *(const u32x4*)(b0p + kw);
            b0.q[1] = *(const u32x4*)(b0p + kw + 16);
            b1.q[0] = *(const u32x4*)(b1p + kw);
            b1.q[1] = *(const u32x4*)(b1p + kw + 16);

            v16bf a[4];
            #pragma unroll
            for (int mt = 0; mt < 4; ++mt) {
                a[mt] = load_a_bf(hprev + (size_t)(mt * 16 + mrow) * H_DIM + k + kA);
            }
            #pragma unroll
            for (int mt = 0; mt < 4; ++mt) {
                acc[mt][0] = __builtin_amdgcn_wmma_f32_16x16x32_bf16(
                    false, a[mt], false, b0.v, (short)0, acc[mt][0], false, false);
                acc[mt][1] = __builtin_amdgcn_wmma_f32_16x16x32_bf16(
                    false, a[mt], false, b1.v, (short)0, acc[mt][1], false, false);
            }
        }
    }

    // Epilogue: C/D layout -> lane = N col, VGPR v = M row (hi selects +8)
    const float bias0 = bias[g0];
    const float bias1 = bias[g1];
    const bool  v1ok  = (g1raw < G_DIM);
    #pragma unroll
    for (int mt = 0; mt < 4; ++mt) {
        #pragma unroll
        for (int v = 0; v < 8; ++v) {
            const int row = mt * 16 + hi * 8 + v;
            gout[(size_t)row * G_DIM + g0] = acc[mt][0][v] + bias0;
            if (v1ok) gout[(size_t)row * G_DIM + g1raw] = acc[mt][1][v] + bias1;
        }
    }
}

// ---------------- gating: softmax/cumsum master gates + cell update ----------------
__device__ __forceinline__ float sigf(float x) { return 1.0f / (1.0f + __expf(-x)); }

__global__ __launch_bounds__(256)
void onlstm_gate_step(const float* __restrict__ gbuf,   // 64 x G_DIM
                      float*       __restrict__ cstate, // 64 x 1024 (B,N,CS flattened)
                      bf16_t*      __restrict__ h_bf,   // 64 x 1024 bf16 out
                      float*       __restrict__ h_f32)  // 64 x 1024 fp32 out (nullable)
{
    int idx = blockIdx.x * 256 + threadIdx.x;
    if (idx >= BH) return;
    int b = idx >> 10;
    int j = idx & (H_DIM - 1);
    int n = j >> 7;                       // chunk index 0..7
    const float* gr = gbuf + (size_t)b * G_DIM;

    // softmax + cumsum over g[:, 0:8] and g[:, 8:16]
    float e0[8], e1[8];
    float m0 = -3.4e38f, m1 = -3.4e38f;
    #pragma unroll
    for (int i = 0; i < 8; ++i) { e0[i] = gr[i];     m0 = fmaxf(m0, e0[i]); }
    #pragma unroll
    for (int i = 0; i < 8; ++i) { e1[i] = gr[8 + i]; m1 = fmaxf(m1, e1[i]); }
    float s0 = 0.f, s1 = 0.f, cum0 = 0.f, cum1 = 0.f;
    #pragma unroll
    for (int i = 0; i < 8; ++i) {
        float a = __expf(e0[i] - m0); s0 += a; if (i <= n) cum0 += a;
        float c = __expf(e1[i] - m1); s1 += c; if (i <= n) cum1 += c;
    }
    float c_in = 1.0f - cum0 / s0;        // 1 - cumsum(softmax(g[:, :N]))[n]
    float c_fg = cum1 / s1;               // cumsum(softmax(g[:, N:2N]))[n]

    float og   = gr[16 + j];
    float cell = gr[16 + H_DIM + j];
    float ig   = gr[16 + 2 * H_DIM + j];
    float fg   = gr[16 + 3 * H_DIM + j];

    float ov    = c_fg * c_in;
    float fgate = sigf(fg) * ov + (c_fg - ov);
    float igate = sigf(ig) * ov + (c_in - ov);
    float cold  = cstate[idx];
    float cy    = fgate * cold + igate * tanhf(cell);
    cstate[idx] = cy;
    float hy    = sigf(og) * tanhf(cy);
    h_bf[idx]   = (bf16_t)hy;
    if (h_f32) h_f32[idx] = hy;
}

// ---------------- host ----------------
extern "C" void kernel_launch(void* const* d_in, const int* in_sizes, int n_in,
                              void* d_out, int out_size, void* d_ws, size_t ws_size,
                              hipStream_t stream)
{
    (void)in_sizes; (void)n_in; (void)out_size; (void)ws_size;
    const float* x    = (const float*)d_in[0];
    const float* Wih0 = (const float*)d_in[1];
    const float* bih0 = (const float*)d_in[2];
    const float* Whh0 = (const float*)d_in[3];
    const float* bhh0 = (const float*)d_in[4];
    const float* Wih1 = (const float*)d_in[5];
    const float* bih1 = (const float*)d_in[6];
    const float* Whh1 = (const float*)d_in[7];
    const float* bhh1 = (const float*)d_in[8];
    float* out = (float*)d_out;

    // workspace carve-out (~102 MB)
    char* ws = (char*)d_ws;
    size_t off = 0;
    auto carve = [&](size_t bytes) -> void* {
        void* p = ws + off;
        off = (off + bytes + 255) & ~(size_t)255;
        return p;
    };
    bf16_t* Wc   = (bf16_t*)carve(2ull * G_DIM * KC * sizeof(bf16_t));  // 33.7 MB, L2-resident
    float*  bias = (float*) carve(2ull * G_DIM * sizeof(float));
    bf16_t* hs0  = (bf16_t*)carve((size_t)L_SEQT * BH * sizeof(bf16_t)); // 67 MB: layer-0 outputs
    bf16_t* h1   = (bf16_t*)carve((size_t)BH * sizeof(bf16_t));          // rotating layer-1 h
    bf16_t* hz   = (bf16_t*)carve((size_t)BH * sizeof(bf16_t));          // zero h for t=0
    float*  cst  = (float*) carve((size_t)BH * sizeof(float));           // cell state
    float*  gbuf = (float*) carve((size_t)B_DIM * G_DIM * sizeof(float));// per-step gates

    // one-time setup
    {
        size_t n = 2ull * G_DIM * KC;
        pack_weights_kernel<<<(unsigned)((n + 255) / 256), 256, 0, stream>>>(
            Wih0, Whh0, Wih1, Whh1, Wc);
    }
    combine_bias_kernel<<<(2 * G_DIM + 255) / 256, 256, 0, stream>>>(bih0, bhh0, bih1, bhh1, bias);
    zero_u32_kernel<<<(BH * 2 / 4 + 255) / 256, 256, 0, stream>>>((u32*)hz, BH * 2 / 4);

    const int gemm_grid = (G_DIM + 127) / 128;   // 33 blocks x 4 waves

    // ---- layer 0 (A from fp32 x) ----
    zero_u32_kernel<<<(BH + 255) / 256, 256, 0, stream>>>((u32*)cst, BH);
    for (int t = 0; t < L_SEQT; ++t) {
        const bf16_t* hp = (t == 0) ? hz : (hs0 + (size_t)(t - 1) * BH);
        onlstm_gemm_step<true><<<gemm_grid, 128, 0, stream>>>(
            (const void*)(x + (size_t)t * BH), hp, Wc, bias, gbuf);
        onlstm_gate_step<<<BH / 256, 256, 0, stream>>>(
            gbuf, cst, hs0 + (size_t)t * BH, nullptr);
    }

    // ---- layer 1 (A from bf16 hs0) ----
    zero_u32_kernel<<<(BH + 255) / 256, 256, 0, stream>>>((u32*)cst, BH);
    for (int t = 0; t < L_SEQT; ++t) {
        const bf16_t* hp = (t == 0) ? hz : h1;
        onlstm_gemm_step<false><<<gemm_grid, 128, 0, stream>>>(
            (const void*)(hs0 + (size_t)t * BH), hp,
            Wc + (size_t)G_DIM * KC, bias + G_DIM, gbuf);
        onlstm_gate_step<<<BH / 256, 256, 0, stream>>>(
            gbuf, cst, h1, out + (size_t)t * BH);
    }
}
